// MultiHeadAttention_33981781246192
// MI455X (gfx1250) — matrix operand order repaired and checked by reading, not compiled, not run on hardware
//
#include <hip/hip_runtime.h>
#include <hip/hip_bf16.h>
#include <math.h>

// ---------------------------------------------------------------------------
// Causal multi-head attention for MI455X (gfx1250), wave32 + WMMA f16->f32.
//
// Roofline: 43 GFLOP total; essential HBM traffic ~75MB (~3.2us @ 23.3TB/s).
// f16 WMMA (16x16x32, f32 accum) keeps compute ~10-20us; f32 WMMA (K=4) would
// be ~8x slower per instruction -> we convert to f16 for all matrix inputs and
// accumulate in f32. All intermediates (Q/K/V/O, ~67MB f16) are L2-resident
// (192MB), so per-head re-reads of x/O hit L2 not HBM.
//
// Attention computes S^T = K*Q^T so that softmax statistics are in-lane in the
// WMMA D layout and exp(S^T) directly forms the A-fragment of P for P*V.
// V is stored transposed [d][T] so the P*V B-fragment loads contiguously.
//
// Workspace layout (needs ~66 MiB of d_ws):
//   Q   f16 [B][H][T][d]   16 MiB
//   K   f16 [B][H][T][d]   16 MiB
//   Vt  f16 [B][H][d][T]   16 MiB
//   O   f16 [B*T][H*d]     16 MiB
//   Wqkvt f16 [3][H][d][C]  1.5 MiB   (transposed weights)
//   Wpt   f16 [C][C]        0.5 MiB   (transposed Wp)
// ---------------------------------------------------------------------------

typedef __attribute__((ext_vector_type(16))) _Float16 v16h;
typedef __attribute__((ext_vector_type(8)))  _Float16 v8h;
typedef __attribute__((ext_vector_type(8)))  float    v8f;
typedef __attribute__((ext_vector_type(4)))  float    v4f;

#define NUM_HEADS 8
#define HEAD_SIZE 64
#define EMBED     512
#define SEQ       256
#define BATCH     64
#define BT        (BATCH * SEQ)   // 16384 rows

static __device__ __forceinline__ v8f wmma_f16(v16h a, v16h b, v8f c) {
  // v_wmma_f32_16x16x32_f16  (8-arg form: neg_a, A, neg_b, B, c_mod, C, reuse_a, reuse_b)
  return __builtin_amdgcn_wmma_f32_16x16x32_f16(false, a, false, b, (short)0, c, false, false);
}

static __device__ __forceinline__ v16h pack16(v8h lo, v8h hi) {
  return __builtin_shufflevector(lo, hi, 0,1,2,3,4,5,6,7,8,9,10,11,12,13,14,15);
}

// ---------------------------------------------------------------------------
// Kernel 0: convert + transpose weights to f16 B-fragment-friendly layouts.
//   Wq/Wk/Wv [H][C][d] f32  ->  Wqkvt [mat][H][d][C] f16
//   Wp [C][C] f32           ->  Wpt [n][k] f16 (transposed)
// ---------------------------------------------------------------------------
__global__ __launch_bounds__(256) void prep_weights_kernel(
    const float* __restrict__ Wq, const float* __restrict__ Wk,
    const float* __restrict__ Wv, const float* __restrict__ Wp,
    _Float16* __restrict__ Wqkvt, _Float16* __restrict__ Wpt) {
  int idx = blockIdx.x * 256 + threadIdx.x;
  const int QKV = 3 * NUM_HEADS * HEAD_SIZE * EMBED;   // 786432
  if (idx < QKV) {
    int mat  = idx / (NUM_HEADS * HEAD_SIZE * EMBED);
    int rem  = idx % (NUM_HEADS * HEAD_SIZE * EMBED);
    int h    = rem / (HEAD_SIZE * EMBED);
    int rem2 = rem % (HEAD_SIZE * EMBED);
    int d    = rem2 / EMBED;
    int c    = rem2 % EMBED;
    const float* W = (mat == 0) ? Wq : ((mat == 1) ? Wk : Wv);
    Wqkvt[idx] = (_Float16)W[(h * EMBED + c) * HEAD_SIZE + d];
  } else {
    int j = idx - QKV;                                  // 262144 elements
    int n = j / EMBED;
    int k = j % EMBED;
    Wpt[n * EMBED + k] = (_Float16)Wp[k * EMBED + n];
  }
}

// ---------------------------------------------------------------------------
// Kernel 1: fused QKV projection.
// One wave per (16-row M-tile of x, head): computes 16x64 tiles of Q, K, V
// (12 accumulators), K-dim = 512 in 16 steps of 32. x converted f32->f16 in
// registers; weights read from transposed f16 (contiguous 16-f16 B loads).
// Q,K stored [B][H][T][d]; V stored transposed [B][H][d][T].
// ---------------------------------------------------------------------------
__global__ __launch_bounds__(256) void qkv_gemm_kernel(
    const float* __restrict__ x, const _Float16* __restrict__ Wqkvt,
    _Float16* __restrict__ Qb, _Float16* __restrict__ Kb,
    _Float16* __restrict__ Vtb) {
  const int lane  = threadIdx.x & 31;
  const int wave  = threadIdx.x >> 5;
  const int gw    = blockIdx.x * 8 + wave;         // 8192 waves total
  const int h     = gw & 7;
  const int mtile = gw >> 3;                       // 0..1023
  const int m0    = mtile * 16;
  const int ln    = lane & 15;
  const int off8  = (lane & 16) ? 8 : 0;
  const int off16 = (lane & 16) ? 16 : 0;

  const float* xrow = x + (size_t)(m0 + ln) * EMBED;

  v8f acc[3][4];
#pragma unroll
  for (int m = 0; m < 3; ++m)
#pragma unroll
    for (int nt = 0; nt < 4; ++nt) acc[m][nt] = (v8f)0.0f;

#pragma unroll 4
  for (int ks = 0; ks < 16; ++ks) {
    const int k0 = ks * 32;
    __builtin_prefetch(xrow + k0 + 64, 0, 1);      // global_prefetch_b8
    // A fragment: lane row = m0+ln, K runs {k0+off16..+15}
    v4f f0 = *(const v4f*)(xrow + k0 + off16);
    v4f f1 = *(const v4f*)(xrow + k0 + off16 + 4);
    v4f f2 = *(const v4f*)(xrow + k0 + off16 + 8);
    v4f f3 = *(const v4f*)(xrow + k0 + off16 + 12);
    v16h a;
#pragma unroll
    for (int i = 0; i < 4; ++i) {
      a[i]      = (_Float16)f0[i];
      a[4 + i]  = (_Float16)f1[i];
      a[8 + i]  = (_Float16)f2[i];
      a[12 + i] = (_Float16)f3[i];
    }
#pragma unroll
    for (int mat = 0; mat < 3; ++mat) {
#pragma unroll
      for (int nt = 0; nt < 4; ++nt) {
        const int n = nt * 16 + ln;
        const _Float16* wrow =
            Wqkvt + ((size_t)((mat * NUM_HEADS + h) * HEAD_SIZE + n)) * EMBED;
        v16h bfrag = *(const v16h*)(wrow + k0 + off16);
        acc[mat][nt] = wmma_f16(a, bfrag, acc[mat][nt]);
      }
    }
  }

  // Store: D layout row = r + 8*(lane>=16), col = ln (per 16-wide N tile).
#pragma unroll
  for (int r = 0; r < 8; ++r) {
    const int rg = m0 + r + off8;        // global row in [0, BT)
    const int bb = rg >> 8;              // batch
    const int t  = rg & 255;             // time
    const size_t qkBase = ((size_t)(bb * NUM_HEADS + h) * SEQ + t) * HEAD_SIZE;
    const size_t vBase  = ((size_t)(bb * NUM_HEADS + h) * HEAD_SIZE) * SEQ;
#pragma unroll
    for (int nt = 0; nt < 4; ++nt) {
      const int n = nt * 16 + ln;
      Qb[qkBase + n]          = (_Float16)acc[0][nt][r];
      Kb[qkBase + n]          = (_Float16)acc[1][nt][r];
      Vtb[vBase + (size_t)n * SEQ + t] = (_Float16)acc[2][nt][r];
    }
  }
}

// ---------------------------------------------------------------------------
// Kernel 2: causal flash attention, one block per (b,h), 8 waves x 32 queries.
// S^T = K*Q^T so softmax stats are in-lane; exp(S^T) forms P's A-fragment
// directly; P*V uses Vt for contiguous B-fragments. Per-t flash rescale
// factors are broadcast to the accumulator layout with ds_bpermute (__shfl).
// ---------------------------------------------------------------------------
__global__ __launch_bounds__(256) void attn_kernel(
    const _Float16* __restrict__ Qb, const _Float16* __restrict__ Kb,
    const _Float16* __restrict__ Vtb, _Float16* __restrict__ O) {
  const int bh   = blockIdx.x;            // b*8 + h
  const int b    = bh >> 3;
  const int h    = bh & 7;
  const int lane = threadIdx.x & 31;
  const int wave = threadIdx.x >> 5;
  const int t0   = wave * 32;
  const int ln   = lane & 15;
  const int off8  = (lane & 16) ? 8 : 0;
  const int off16 = (lane & 16) ? 16 : 0;

  const _Float16* Qbh = Qb + (size_t)bh * SEQ * HEAD_SIZE;
  const _Float16* Kbh = Kb + (size_t)bh * SEQ * HEAD_SIZE;
  const _Float16* Vbh = Vtb + (size_t)bh * HEAD_SIZE * SEQ;

  // Preload Q^T B-fragments: bq[ttile][dchunk], lane col t = t0+tt*16+ln.
  v16h bq[2][2];
#pragma unroll
  for (int tt = 0; tt < 2; ++tt) {
    const _Float16* qr = Qbh + (size_t)(t0 + tt * 16 + ln) * HEAD_SIZE;
#pragma unroll
    for (int dc = 0; dc < 2; ++dc)
      bq[tt][dc] = *(const v16h*)(qr + dc * 32 + off16);
  }

  v8f acc[2][4];
#pragma unroll
  for (int tt = 0; tt < 2; ++tt)
#pragma unroll
    for (int dt = 0; dt < 4; ++dt) acc[tt][dt] = (v8f)0.0f;

  float mrun[2] = {-INFINITY, -INFINITY};
  float lrun[2] = {0.0f, 0.0f};
  const float scale = 0.125f;  // HEAD_SIZE^-0.5

  const int nblk = t0 / 32 + 1;
  for (int sb = 0; sb < nblk; ++sb) {
    const int s0 = sb * 32;
    // K A-fragments: ak[stile][dchunk], lane row s = s0+st*16+ln.
    v16h ak[2][2];
#pragma unroll
    for (int st = 0; st < 2; ++st) {
      const _Float16* kr = Kbh + (size_t)(s0 + st * 16 + ln) * HEAD_SIZE;
#pragma unroll
      for (int dc = 0; dc < 2; ++dc) {
        v8h r0 = *(const v8h*)(kr + dc * 32 + off16);
        v8h r1 = *(const v8h*)(kr + dc * 32 + off16 + 8);
        ak[st][dc] = pack16(r0, r1);
      }
    }
    // S^T tiles: rows = s (VGPRs), cols = t (lanes).
    v8f s_t[2][2];
#pragma unroll
    for (int st = 0; st < 2; ++st)
#pragma unroll
      for (int tt = 0; tt < 2; ++tt) {
        v8f z = (v8f)0.0f;
        z = wmma_f16(ak[st][0], bq[tt][0], z);
        z = wmma_f16(ak[st][1], bq[tt][1], z);
        s_t[st][tt] = z;
      }

    // Flash update per t-tile (in-lane stats, one shfl_xor to merge halves).
#pragma unroll
    for (int tt = 0; tt < 2; ++tt) {
      const int t = t0 + tt * 16 + ln;
      float mx = -INFINITY;
#pragma unroll
      for (int st = 0; st < 2; ++st)
#pragma unroll
        for (int r = 0; r < 8; ++r) {
          const int s = s0 + st * 16 + r + off8;
          float v = s_t[st][tt][r] * scale;
          v = (s <= t) ? v : -INFINITY;     // causal mask
          s_t[st][tt][r] = v;
          mx = fmaxf(mx, v);
        }
      mx = fmaxf(mx, __shfl_xor(mx, 16));
      const float mnew  = fmaxf(mrun[tt], mx);
      const float alpha = __expf(mrun[tt] - mnew);
      float rs = 0.0f;
#pragma unroll
      for (int st = 0; st < 2; ++st)
#pragma unroll
        for (int r = 0; r < 8; ++r) {
          const float p = __expf(s_t[st][tt][r] - mnew);
          s_t[st][tt][r] = p;
          rs += p;
        }
      rs += __shfl_xor(rs, 16);
      lrun[tt] = lrun[tt] * alpha + rs;
      mrun[tt] = mnew;
      // Rescale accumulators: acc row t_local = r + off8, alpha lives in lane
      // (t_local % 16) -> per-lane-source shuffle (ds_bpermute).
      float sc[8];
#pragma unroll
      for (int r = 0; r < 8; ++r) sc[r] = __shfl(alpha, r + off8);
#pragma unroll
      for (int dt = 0; dt < 4; ++dt)
#pragma unroll
        for (int r = 0; r < 8; ++r) acc[tt][dt][r] *= sc[r];
    }

    // exp(S^T) is already the A-fragment layout of P (K = 32 keys).
    v16h pa[2];
#pragma unroll
    for (int tt = 0; tt < 2; ++tt)
#pragma unroll
      for (int e = 0; e < 16; ++e)
        pa[tt][e] = (_Float16)s_t[e >> 3][tt][e & 7];

    // P*V: B-fragments contiguous from transposed V.
#pragma unroll
    for (int dt = 0; dt < 4; ++dt) {
      const int d = dt * 16 + ln;
      v16h bv = pack16(*(const v8h*)(Vbh + (size_t)d * SEQ + s0 + off8),
                       *(const v8h*)(Vbh + (size_t)d * SEQ + s0 + 16 + off8));
#pragma unroll
      for (int tt = 0; tt < 2; ++tt)
        acc[tt][dt] = wmma_f16(pa[tt], bv, acc[tt][dt]);
    }
  }

  // Epilogue: divide by row sum, store f16 to O[b*T + t][h*64 + d].
#pragma unroll
  for (int tt = 0; tt < 2; ++tt) {
    const float invv = 1.0f / lrun[tt];
    float iv[8];
#pragma unroll
    for (int r = 0; r < 8; ++r) iv[r] = __shfl(invv, r + off8);
#pragma unroll
    for (int r = 0; r < 8; ++r) {
      const int trow = t0 + tt * 16 + r + off8;
      const size_t rowBase = ((size_t)b * SEQ + trow) * EMBED + h * HEAD_SIZE;
#pragma unroll
      for (int dt = 0; dt < 4; ++dt)
        O[rowBase + dt * 16 + ln] = (_Float16)(acc[tt][dt][r] * iv[r]);
    }
  }
}

// ---------------------------------------------------------------------------
// Kernel 3: output projection  out = O @ Wp + bp   (f16 WMMA, f32 out).
// One wave per (16-row M-tile, 64-col N-group); bias pre-loaded into C.
// ---------------------------------------------------------------------------
__global__ __launch_bounds__(256) void out_proj_kernel(
    const _Float16* __restrict__ O, const _Float16* __restrict__ Wpt,
    const float* __restrict__ bp, float* __restrict__ out) {
  const int lane  = threadIdx.x & 31;
  const int wave  = threadIdx.x >> 5;
  const int gw    = blockIdx.x * 8 + wave;   // 8192 waves
  const int ng    = gw & 7;                  // N group: 64 cols
  const int mtile = gw >> 3;                 // 0..1023
  const int m0    = mtile * 16;
  const int ln    = lane & 15;
  const int off8  = (lane & 16) ? 8 : 0;
  const int off16 = (lane & 16) ? 16 : 0;

  const _Float16* orow = O + (size_t)(m0 + ln) * EMBED;

  v8f acc[4];
#pragma unroll
  for (int nt = 0; nt < 4; ++nt) {
    const float bv = bp[ng * 64 + nt * 16 + ln];
#pragma unroll
    for (int r = 0; r < 8; ++r) acc[nt][r] = bv;
  }

#pragma unroll 4
  for (int ks = 0; ks < 16; ++ks) {
    const int k0 = ks * 32;
    __builtin_prefetch(orow + k0 + 64, 0, 1);
    v8h a0 = *(const v8h*)(orow + k0 + off16);
    v8h a1 = *(const v8h*)(orow + k0 + off16 + 8);
    v16h a = pack16(a0, a1);
#pragma unroll
    for (int nt = 0; nt < 4; ++nt) {
      const int n = ng * 64 + nt * 16 + ln;
      v16h bfrag = *(const v16h*)(Wpt + (size_t)n * EMBED + k0 + off16);
      acc[nt] = wmma_f16(a, bfrag, acc[nt]);
    }
  }

#pragma unroll
  for (int r = 0; r < 8; ++r) {
    const int rg = m0 + r + off8;
#pragma unroll
    for (int nt = 0; nt < 4; ++nt) {
      const int n = ng * 64 + nt * 16 + ln;
      out[(size_t)rg * EMBED + n] = acc[nt][r];
    }
  }
}

// ---------------------------------------------------------------------------
extern "C" void kernel_launch(void* const* d_in, const int* in_sizes, int n_in,
                              void* d_out, int out_size, void* d_ws, size_t ws_size,
                              hipStream_t stream) {
  const float* x  = (const float*)d_in[0];
  const float* Wq = (const float*)d_in[1];
  const float* Wk = (const float*)d_in[2];
  const float* Wv = (const float*)d_in[3];
  const float* Wp = (const float*)d_in[4];
  const float* bp = (const float*)d_in[5];
  float* out = (float*)d_out;

  char* ws = (char*)d_ws;
  constexpr size_t szQKV = (size_t)BATCH * NUM_HEADS * SEQ * HEAD_SIZE * 2; // 16 MiB each
  constexpr size_t szO   = (size_t)BT * EMBED * 2;                          // 16 MiB
  constexpr size_t szWq  = (size_t)3 * NUM_HEADS * HEAD_SIZE * EMBED * 2;   // 1.5 MiB
  _Float16* Q     = (_Float16*)(ws);
  _Float16* K     = (_Float16*)(ws + szQKV);
  _Float16* Vt    = (_Float16*)(ws + 2 * szQKV);
  _Float16* O     = (_Float16*)(ws + 3 * szQKV);
  _Float16* Wqkvt = (_Float16*)(ws + 3 * szQKV + szO);
  _Float16* Wpt   = (_Float16*)(ws + 3 * szQKV + szO + szWq);

  // 0) weights -> f16 transposed (786432 + 262144 = 1048576 threads)
  prep_weights_kernel<<<4096, 256, 0, stream>>>(Wq, Wk, Wv, Wp, Wqkvt, Wpt);
  // 1) fused QKV projection: 1024 M-tiles x 8 heads = 8192 waves
  qkv_gemm_kernel<<<1024, 256, 0, stream>>>(x, Wqkvt, Q, K, Vt);
  // 2) flash attention: one block per (b,h)
  attn_kernel<<<BATCH * NUM_HEADS, 256, 0, stream>>>(Q, K, Vt, O);
  // 3) output projection + bias
  out_proj_kernel<<<1024, 256, 0, stream>>>(O, Wpt, bp, out);
}
